// MultiLinearLayer_39908836114668
// MI455X (gfx1250) — matrix operand order repaired and checked by reading, not compiled, not run on hardware
//
#include <hip/hip_runtime.h>
#include <stdint.h>

// ---------------------------------------------------------------------------
// Problem constants (from reference)
// ---------------------------------------------------------------------------
#define IN_F   1024
#define OUT_F  1024
#define NROW   16384
#define NTERMS 10          // GRID + ORDER

typedef __attribute__((ext_vector_type(16))) __bf16 v16bf;
typedef __attribute__((ext_vector_type(8)))  float  v8f;
typedef __attribute__((ext_vector_type(4)))  int    v4i;

#ifndef __has_builtin
#define __has_builtin(x) 0
#endif

#if __has_builtin(__builtin_amdgcn_global_load_async_to_lds_b128)
#define HAVE_ASYNC 1
#else
#define HAVE_ASYNC 0
#endif

// exact pointer types expected by the async builtin (from hipcc diagnostic:
// param 0 is 'int4 __device__ *', i.e. non-const AS(1) v4i*)
typedef __attribute__((address_space(1))) v4i* gp128;
typedef __attribute__((address_space(3))) v4i* lp128;

// fp32 -> bf16 round-to-nearest-even
__device__ __forceinline__ unsigned int f2bf1(float f) {
    unsigned int u = __float_as_uint(f);
    return (u + 0x7FFFu + ((u >> 16) & 1u)) >> 16;
}
__device__ __forceinline__ unsigned int pk2(float a, float b) {
    return f2bf1(a) | (f2bf1(b) << 16);
}

// 16-byte global -> LDS copy: async DMA if available, else sync load+store
__device__ __forceinline__ void copy16(const unsigned short* g, unsigned short* l) {
#if HAVE_ASYNC
    __builtin_amdgcn_global_load_async_to_lds_b128(
        (gp128)(uintptr_t)g, (lp128)(unsigned int)(uintptr_t)l, 0, 0);
#else
    *(uint4*)l = *(const uint4*)g;
#endif
}

template <int N>
__device__ __forceinline__ void wait_async() {
#if HAVE_ASYNC
#if __has_builtin(__builtin_amdgcn_s_wait_asynccnt)
    __builtin_amdgcn_s_wait_asynccnt(N);
#else
    if (N == 0) asm volatile("s_wait_asynccnt 0x0" ::: "memory");
    else        asm volatile("s_wait_asynccnt 0x8" ::: "memory");
#endif
#endif
}

// two ds_load_b128 -> one 16-element bf16 WMMA operand
__device__ __forceinline__ v16bf load_frag(const unsigned short* p0,
                                           const unsigned short* p1) {
    union { uint4 u[2]; v16bf v; } c;
    c.u[0] = *(const uint4*)p0;
    c.u[1] = *(const uint4*)p1;
    return c.v;
}

// ---------------------------------------------------------------------------
// Prepack kernels: everything the GEMM touches becomes bf16 in d_ws
// ---------------------------------------------------------------------------
__global__ void pack_x_kernel(const float* __restrict__ x,
                              unsigned short* __restrict__ xb) {
    size_t g = ((size_t)blockIdx.x * 256 + threadIdx.x) * 8;
    float4 f0 = *(const float4*)(x + g);
    float4 f1 = *(const float4*)(x + g + 4);
    uint4 o;
    o.x = pk2(f0.x, f0.y);
    o.y = pk2(f0.z, f0.w);
    o.z = pk2(f1.x, f1.y);
    o.w = pk2(f1.z, f1.w);
    *(uint4*)(xb + g) = o;
}

__global__ void pack_base_kernel(const float* __restrict__ bw,
                                 unsigned short* __restrict__ wb) {
    int g = (blockIdx.x * 256 + threadIdx.x) * 4;
    float4 f = *(const float4*)(bw + g);
    uint2 o;
    o.x = pk2(f.x, f.y);
    o.y = pk2(f.z, f.w);
    *(uint2*)(wb + g) = o;
}

__global__ void pack_spline_kernel(const float* __restrict__ sp,
                                   unsigned short* __restrict__ ws) {
    int g = blockIdx.x * 256 + threadIdx.x;   // 0 .. 1M-1
    const float* p = sp + (size_t)g * NTERMS;
    float s = 0.f;
#pragma unroll
    for (int t = 0; t < NTERMS; ++t) s += p[t];
    ws[g] = (unsigned short)f2bf1(s * (1.0f / NTERMS));
}

// ---------------------------------------------------------------------------
// Fused dual-GEMM: out = silu(x @ Wb^T) + x @ Ws^T      (all operands bf16)
//   block: 256 threads = 8 wave32, tile 128(M) x 64(N), BK = 64
//   waves: 4 along M (32 rows each) x 2 along N (32 cols each)
//   LDS double-buffered, filled by async load-to-LDS (ASYNCcnt pipelined)
// ---------------------------------------------------------------------------
#define BK       64
#define LDSTR    72                    // padded LDS row stride (bf16 elems)
#define KTILES   (IN_F / BK)
#define A_ELEMS  (128 * LDSTR)         // 9216
#define B_ELEMS  (64 * LDSTR)          // 4608
#define BUFELEMS (A_ELEMS + 2 * B_ELEMS)   // 18432 elems = 36864 B per buffer

__global__ __launch_bounds__(256)
void kan_gemm_kernel(const unsigned short* __restrict__ Xb,
                     const unsigned short* __restrict__ Wb,
                     const unsigned short* __restrict__ Ws,
                     float* __restrict__ Out) {
    __shared__ __align__(16) unsigned short lds[2 * BUFELEMS];

    const int tid   = threadIdx.x;
    const int lane  = tid & 31;
    const int wave  = tid >> 5;
    const int waveM = wave & 3;           // 0..3
    const int waveN = wave >> 2;          // 0..1

    const int blockM = blockIdx.y * 128;
    const int blockN = blockIdx.x * 64;

    // tile copy: 16-byte units. A: 1024 units (4/thread), B: 512 units (2/thread)
    auto issue_tile = [&](int k0, int buf) {
        const int base = buf * BUFELEMS;
#pragma unroll
        for (int i = 0; i < 4; ++i) {
            const int unit = tid + i * 256;
            const int row  = unit >> 3;
            const int ch   = (unit & 7) * 8;        // bf16 elems
            copy16(Xb + (size_t)(blockM + row) * IN_F + k0 + ch,
                   &lds[base + row * LDSTR + ch]);
        }
#pragma unroll
        for (int i = 0; i < 2; ++i) {
            const int unit = tid + i * 256;
            const int row  = unit >> 3;
            const int ch   = (unit & 7) * 8;
            copy16(Wb + (size_t)(blockN + row) * IN_F + k0 + ch,
                   &lds[base + A_ELEMS + row * LDSTR + ch]);
            copy16(Ws + (size_t)(blockN + row) * IN_F + k0 + ch,
                   &lds[base + A_ELEMS + B_ELEMS + row * LDSTR + ch]);
        }
    };

    v8f zero = {0.f, 0.f, 0.f, 0.f, 0.f, 0.f, 0.f, 0.f};
    v8f accB[2][2], accS[2][2];
#pragma unroll
    for (int i = 0; i < 2; ++i)
#pragma unroll
        for (int j = 0; j < 2; ++j) { accB[i][j] = zero; accS[i][j] = zero; }

    // per-lane WMMA fragment addressing (16-bit A / B, same contiguous K order for both)
    const int mrow  = lane & 15;
    const int aKoff = (lane < 16) ? 0 : 16;   // A: 16 contiguous K at 0 / 16
    const int bKoff = (lane < 16) ? 0 : 16;   // B: 16 contiguous K at 0 / 16

    issue_tile(0, 0);

    for (int t = 0; t < KTILES; ++t) {
        const int buf = t & 1;
        if (t + 1 < KTILES) {
            issue_tile((t + 1) * BK, buf ^ 1);   // 8 more async ops in flight
            wait_async<8>();                     // tile t complete (in-order)
        } else {
            wait_async<0>();
        }
        __syncthreads();                         // tile t visible to all waves

        const unsigned short* As_ = &lds[buf * BUFELEMS];
        const unsigned short* Bb_ = As_ + A_ELEMS;
        const unsigned short* Bs_ = Bb_ + B_ELEMS;

#pragma unroll
        for (int kk = 0; kk < BK; kk += 32) {
            v16bf a[2], bB[2], bS[2];
#pragma unroll
            for (int tm = 0; tm < 2; ++tm) {
                const unsigned short* p =
                    As_ + (waveM * 32 + tm * 16 + mrow) * LDSTR + kk + aKoff;
                a[tm] = load_frag(p, p + 8);
            }
#pragma unroll
            for (int tn = 0; tn < 2; ++tn) {
                const unsigned short* pb =
                    Bb_ + (waveN * 32 + tn * 16 + mrow) * LDSTR + kk + bKoff;
                bB[tn] = load_frag(pb, pb + 8);
                const unsigned short* ps =
                    Bs_ + (waveN * 32 + tn * 16 + mrow) * LDSTR + kk + bKoff;
                bS[tn] = load_frag(ps, ps + 8);
            }
#pragma unroll
            for (int tm = 0; tm < 2; ++tm)
#pragma unroll
                for (int tn = 0; tn < 2; ++tn) {
                    accB[tm][tn] = __builtin_amdgcn_wmma_f32_16x16x32_bf16(
                        false, a[tm], false, bB[tn], (short)0, accB[tm][tn],
                        false, false);
                    accS[tm][tn] = __builtin_amdgcn_wmma_f32_16x16x32_bf16(
                        false, a[tm], false, bS[tn], (short)0, accS[tm][tn],
                        false, false);
                }
        }
        __syncthreads();   // all waves done reading buf before it is re-filled
    }

    // epilogue: out = silu(base) + spline  (C layout: VGPR r -> row r, +8 hi lanes)
    const int rowHi = (lane >> 4) << 3;
#pragma unroll
    for (int tm = 0; tm < 2; ++tm)
#pragma unroll
        for (int tn = 0; tn < 2; ++tn) {
            const int col  = blockN + waveN * 32 + tn * 16 + mrow;
            const int row0 = blockM + waveM * 32 + tm * 16 + rowHi;
#pragma unroll
            for (int r = 0; r < 8; ++r) {
                float b = accB[tm][tn][r];
                float s = accS[tm][tn][r];
                float v = b / (1.0f + __expf(-b)) + s;
                Out[(size_t)(row0 + r) * OUT_F + col] = v;
            }
        }
}

// ---------------------------------------------------------------------------
// launch
// ---------------------------------------------------------------------------
extern "C" void kernel_launch(void* const* d_in, const int* in_sizes, int n_in,
                              void* d_out, int out_size, void* d_ws, size_t ws_size,
                              hipStream_t stream) {
    const float* x  = (const float*)d_in[0];
    const float* bw = (const float*)d_in[1];
    const float* sp = (const float*)d_in[2];
    float* out = (float*)d_out;

    unsigned short* Xbf = (unsigned short*)d_ws;                  // 32 MB
    unsigned short* Wb  = Xbf + (size_t)NROW * IN_F;              //  2 MB
    unsigned short* Ws  = Wb + (size_t)OUT_F * IN_F;              //  2 MB

    pack_x_kernel<<<(NROW * IN_F) / (256 * 8), 256, 0, stream>>>(x, Xbf);
    pack_base_kernel<<<(OUT_F * IN_F) / (256 * 4), 256, 0, stream>>>(bw, Wb);
    pack_spline_kernel<<<(OUT_F * IN_F) / 256, 256, 0, stream>>>(sp, Ws);

    dim3 grid(OUT_F / 64, NROW / 128);
    kan_gemm_kernel<<<grid, 256, 0, stream>>>(Xbf, Wb, Ws, out);
}
